// RNN_stack_57191784514175
// MI455X (gfx1250) — matrix operand order repaired and checked by reading, not compiled, not run on hardware
//
#include <hip/hip_runtime.h>
#include <stdint.h>

// ---------------------------------------------------------------------------
// Stacked leaky-ReLU RNN for MI455X (gfx1250), wave32 + WMMA bf16.
//
// One workgroup (512 threads = 16 waves) owns 16 batch rows and the full
// N=1024 state of all 3 layers across all T=256 steps (recurrence is
// independent per batch row -> zero grid-wide sync). h lives in LDS twice:
// f32 (exact decay term) + bf16 (zero-conversion WMMA A-fragments). Weights
// stream from L2 as bf16. Each wave owns 4 column tiles: one A-fragment
// feeds 4 independent WMMA chains; first K-step of each GEMM is peeled so
// the x-projection accumulator is consumed as the WMMA C operand (VDST !=
// SRC2). W_in fragment loads are kept inside the t-loop via an asm LICM
// barrier so they are re-fetched from L2 instead of spilled to scratch.
// ---------------------------------------------------------------------------

#define T_STEPS   256
#define BATCH     256
#define IN_DIM    64
#define N_DIM     1024
#define N_CLASSES 10
#define BROWS     16            // batch rows per workgroup
#define LDSF      1028          // f32 row stride (bank-conflict break)
#define LDSB      1032          // bf16 row stride (bank-conflict break)
#define NTHREADS  512
#define TILES     4             // column tiles per wave
#define DECAY     0.5f          // 1 - 1/TAU, TAU = 2
#define GAIN      0.5f          // 1/TAU
#define SLOPE     0.01f         // leaky relu

typedef __attribute__((ext_vector_type(16))) __bf16 v16bf;
typedef __attribute__((ext_vector_type(8)))  __bf16 v8bf;
typedef __attribute__((ext_vector_type(16))) float  v16f;
typedef __attribute__((ext_vector_type(8)))  float  v8f;

__device__ __forceinline__ unsigned short f32_to_bf16_bits(float f) {
  union { float f; unsigned u; } v; v.f = f;
  unsigned r = v.u + 0x7FFFu + ((v.u >> 16) & 1u);   // round-to-nearest-even
  return (unsigned short)(r >> 16);
}

// WMMA: D = A(16x32 bf16) x B(32x16 bf16) + C(16x16 f32)
#define WMMA_BF16(a, b, c) \
  __builtin_amdgcn_wmma_f32_16x16x32_bf16(false, (a), false, (b), (short)0, (c), false, false)

// A-fragment (16x32, MxK) straight from the bf16 h copy in LDS: two 16B
// contiguous loads per lane, no conversions.  (ISA 7.12.2, 16-bit A layout:
// lane L<16 = row L, elems[0..7]=K k0..k0+7, elems[8..15]=K k0+16..k0+23;
// lanes 16..31 take the +8-shifted K halves.)
__device__ __forceinline__ v16bf load_a_bf(const unsigned short* hb, int lane, int k0) {
  const int row = lane & 15;
  const int off = (lane & 16) ? 8 : 0;
  const unsigned short* p = hb + row * LDSB + k0 + off;
  v8bf lo = *(const v8bf*)(p);
  v8bf hi = *(const v8bf*)(p + 16);
  return __builtin_shufflevector(lo, hi, 0, 1, 2, 3, 4, 5, 6, 7,
                                 8, 9, 10, 11, 12, 13, 14, 15);
}

// A-fragment from an f32 row-major matrix (global data), converting to bf16.
__device__ __forceinline__ v16bf load_a_f32(const float* base, int ldk, int lane, int k0) {
  const int row = lane & 15;
  const int off = (lane & 16) ? 8 : 0;
  const float* p = base + (size_t)row * ldk + k0 + off;
  float4 f0 = *(const float4*)(p);
  float4 f1 = *(const float4*)(p + 4);
  float4 f2 = *(const float4*)(p + 16);
  float4 f3 = *(const float4*)(p + 20);
  v16f t;
  t[0]=f0.x;  t[1]=f0.y;  t[2]=f0.z;  t[3]=f0.w;
  t[4]=f1.x;  t[5]=f1.y;  t[6]=f1.z;  t[7]=f1.w;
  t[8]=f2.x;  t[9]=f2.y;  t[10]=f2.z; t[11]=f2.w;
  t[12]=f3.x; t[13]=f3.y; t[14]=f3.z; t[15]=f3.w;
  return __builtin_convertvector(t, v16bf);           // v_cvt_pk_bf16_f32
}

// B-fragment (32x16, KxN) of W^T where W is [N][ldk] row-major bf16:
// B[k][n] = W[n0+n][k0+k].  Lane n<16 = col n holds K k0..k0+7, k0+16..k0+23;
// lanes 16..31 hold K k0+8..k0+15, k0+24..k0+31.  Two 16B loads per lane.
__device__ __forceinline__ v16bf load_b16(const unsigned short* W, int ldk,
                                          int lane, int n0, int k0) {
  const int coln = n0 + (lane & 15);
  const int koff = (lane & 16) ? 8 : 0;
  const unsigned short* p = W + (size_t)coln * ldk + k0 + koff;
  v8bf lo = *(const v8bf*)(p);                        // 16-byte aligned
  v8bf hi = *(const v8bf*)(p + 16);
  return __builtin_shufflevector(lo, hi, 0, 1, 2, 3, 4, 5, 6, 7,
                                 8, 9, 10, 11, 12, 13, 14, 15);
}

// Elementwise recurrence update for one 16x16 tile; C/D layout:
// VGPR v -> M = v (+8 for lanes>=16), N = lane&15.  Writes f32 + bf16 copies.
__device__ __forceinline__ void update_h(float* h, unsigned short* hb, int lane,
                                         int n0, const v8f& c, float bias) {
  const int mbase = (lane & 16) ? 8 : 0;
  const int n = n0 + (lane & 15);
#pragma unroll
  for (int v = 0; v < 8; ++v) {
    const int m = mbase + v;
    float val = DECAY * h[m * LDSF + n] + (c[v] + bias) * GAIN;
    val = (val > 0.f) ? val : SLOPE * val;
    h[m * LDSF + n] = val;
    hb[m * LDSB + n] = f32_to_bf16_bits(val);
  }
}

// ---------------------------------------------------------------------------
// Prep: convert weights to bf16 in d_ws; zero diagonal of the recurrent W_hh*.
// ws layout (elements): [A0 1M][A1 1M][A2 1M][Whi0 1M][Whi1 1M][Win 64K]
// ---------------------------------------------------------------------------
__global__ void prep_weights_bf16(const float* __restrict__ Whh0,
                                  const float* __restrict__ Whh1,
                                  const float* __restrict__ Whh2,
                                  const float* __restrict__ Whi0,
                                  const float* __restrict__ Whi1,
                                  const float* __restrict__ Win,
                                  unsigned short* __restrict__ ws) {
  const long NN = (long)N_DIM * N_DIM;
  const long total = 5 * NN + (long)N_DIM * IN_DIM;
  long i = (long)blockIdx.x * blockDim.x + threadIdx.x;
  if (i >= total) return;
  float val;
  if (i < 5 * NN) {
    int  m = (int)(i / NN);
    long r = i - (long)m * NN;
    const float* src = (m == 0) ? Whh0 : (m == 1) ? Whh1 : (m == 2) ? Whh2
                       : (m == 3) ? Whi0 : Whi1;
    val = src[r];
    if (m < 3) {                       // torch fill_diagonal_(0.)
      int row = (int)(r / N_DIM), col = (int)(r - (long)row * N_DIM);
      if (row == col) val = 0.f;
    }
  } else {
    val = Win[i - 5 * NN];
  }
  ws[i] = f32_to_bf16_bits(val);
}

// ---------------------------------------------------------------------------
// Main persistent kernel: one workgroup per 16 batch rows, all T steps.
// ---------------------------------------------------------------------------
__global__ void __launch_bounds__(NTHREADS)
rnn_stack_persist(const float* __restrict__ data,
                  const unsigned short* __restrict__ wbf,
                  const float* __restrict__ b_in,
                  const float* __restrict__ b_hh0,
                  const float* __restrict__ b_hh1,
                  const float* __restrict__ b_hh2,
                  const float* __restrict__ b_hi0,
                  const float* __restrict__ b_hi1,
                  const float* __restrict__ W_fc,
                  const float* __restrict__ b_fc,
                  float* __restrict__ out_h,
                  float* __restrict__ out_fc) {
  extern __shared__ char smem_raw[];
  float* hf = (float*)smem_raw;                        // 3 x [16][LDSF] f32
  unsigned short* hbase =
      (unsigned short*)(smem_raw + (size_t)3 * BROWS * LDSF * sizeof(float));
  float* h0 = hf;
  float* h1 = h0 + BROWS * LDSF;
  float* h2 = h1 + BROWS * LDSF;
  unsigned short* hb0 = hbase;                         // 3 x [16][LDSB] bf16
  unsigned short* hb1 = hb0 + BROWS * LDSB;
  unsigned short* hb2 = hb1 + BROWS * LDSB;

  const int tid  = threadIdx.x;
  const int lane = tid & 31;
  const int wave = tid >> 5;                           // 0..15
  const int b0   = blockIdx.x * BROWS;

  const size_t NN = (size_t)N_DIM * N_DIM;
  const unsigned short* A0  = wbf;
  const unsigned short* A1  = wbf + 1 * NN;
  const unsigned short* A2  = wbf + 2 * NN;
  const unsigned short* Hi0 = wbf + 3 * NN;
  const unsigned short* Hi1 = wbf + 4 * NN;
  const unsigned short* Wi  = wbf + 5 * NN;

  // zero-init hidden state (both copies)
  for (int i = tid; i < 3 * BROWS * LDSF; i += NTHREADS) hf[i] = 0.f;
  for (int i = tid; i < 3 * BROWS * LDSB; i += NTHREADS) hbase[i] = 0;
  __syncthreads();

  // each wave owns 4 column tiles: n0 = wave*64 + 16*j
  const int nbase = wave * (16 * TILES);
  const int col   = lane & 15;

  // per-lane column biases; b_in folded in (x accumulator starts from 0)
  float bt0[TILES], bt1[TILES], bt2[TILES];
#pragma unroll
  for (int j = 0; j < TILES; ++j) {
    const int n = nbase + 16 * j + col;
    const float bi = b_in[n];
    bt0[j] = b_hh0[n] + bi;
    bt1[j] = b_hh1[n] + b_hi0[n] + bi;
    bt2[j] = b_hh2[n] + b_hi1[n] + bi;
  }

  for (int t = 0; t < T_STEPS; ++t) {
    // LICM barrier: keep the (t-invariant) W_in fragment loads inside the
    // t-loop -- re-fetching 8 x 32B from L2 each step is far cheaper than
    // pinning 64 VGPRs (which previously spilled to scratch).
    const unsigned short* WiT = Wi;
    asm volatile("" : "+r"(WiT));

    // ---- xmm = data_t @ W_in^T  (K=64, peeled; kept in regs for all layers)
    const float* dt = data + ((size_t)t * BATCH + b0) * IN_DIM;
    v8f xacc[TILES];
    {
      const v8f zc = {};                    // inline 0 C operand
      v16bf a = load_a_f32(dt, IN_DIM, lane, 0);
#pragma unroll
      for (int j = 0; j < TILES; ++j)
        xacc[j] = WMMA_BF16(a, load_b16(WiT, IN_DIM, lane, nbase + 16 * j, 0), zc);
      a = load_a_f32(dt, IN_DIM, lane, 32);
#pragma unroll
      for (int j = 0; j < TILES; ++j)
        xacc[j] = WMMA_BF16(a, load_b16(WiT, IN_DIM, lane, nbase + 16 * j, 32), xacc[j]);
    }

    v8f acc[TILES];

    // ---- layer 0: h0 = lrelu(decay*h0 + (h0 @ A0^T + b_hh0 + x_t)*g)
    {
      v16bf a = load_a_bf(hb0, lane, 0);    // peel: D = A*B + xacc (no copies)
#pragma unroll
      for (int j = 0; j < TILES; ++j)
        acc[j] = WMMA_BF16(a, load_b16(A0, N_DIM, lane, nbase + 16 * j, 0), xacc[j]);
    }
    for (int k0 = 32; k0 < N_DIM; k0 += 32) {
      v16bf a = load_a_bf(hb0, lane, k0);
#pragma unroll
      for (int j = 0; j < TILES; ++j)
        acc[j] = WMMA_BF16(a, load_b16(A0, N_DIM, lane, nbase + 16 * j, k0), acc[j]);
    }
    __syncthreads();                        // all reads of h0 done
#pragma unroll
    for (int j = 0; j < TILES; ++j)
      update_h(h0, hb0, lane, nbase + 16 * j, acc[j], bt0[j]);
    __syncthreads();                        // h0_new visible

    // ---- layer 1: uses h1 (prev) and h0 (new)
    {
      v16bf a = load_a_bf(hb1, lane, 0);
#pragma unroll
      for (int j = 0; j < TILES; ++j)
        acc[j] = WMMA_BF16(a, load_b16(A1, N_DIM, lane, nbase + 16 * j, 0), xacc[j]);
    }
    for (int k0 = 32; k0 < N_DIM; k0 += 32) {
      v16bf a = load_a_bf(hb1, lane, k0);
#pragma unroll
      for (int j = 0; j < TILES; ++j)
        acc[j] = WMMA_BF16(a, load_b16(A1, N_DIM, lane, nbase + 16 * j, k0), acc[j]);
    }
    for (int k0 = 0; k0 < N_DIM; k0 += 32) {
      v16bf a = load_a_bf(hb0, lane, k0);
#pragma unroll
      for (int j = 0; j < TILES; ++j)
        acc[j] = WMMA_BF16(a, load_b16(Hi0, N_DIM, lane, nbase + 16 * j, k0), acc[j]);
    }
    __syncthreads();
#pragma unroll
    for (int j = 0; j < TILES; ++j)
      update_h(h1, hb1, lane, nbase + 16 * j, acc[j], bt1[j]);
    __syncthreads();

    // ---- layer 2: uses h2 (prev) and h1 (new)
    {
      v16bf a = load_a_bf(hb2, lane, 0);
#pragma unroll
      for (int j = 0; j < TILES; ++j)
        acc[j] = WMMA_BF16(a, load_b16(A2, N_DIM, lane, nbase + 16 * j, 0), xacc[j]);
    }
    for (int k0 = 32; k0 < N_DIM; k0 += 32) {
      v16bf a = load_a_bf(hb2, lane, k0);
#pragma unroll
      for (int j = 0; j < TILES; ++j)
        acc[j] = WMMA_BF16(a, load_b16(A2, N_DIM, lane, nbase + 16 * j, k0), acc[j]);
    }
    for (int k0 = 0; k0 < N_DIM; k0 += 32) {
      v16bf a = load_a_bf(hb1, lane, k0);
#pragma unroll
      for (int j = 0; j < TILES; ++j)
        acc[j] = WMMA_BF16(a, load_b16(Hi1, N_DIM, lane, nbase + 16 * j, k0), acc[j]);
    }
    __syncthreads();
#pragma unroll
    for (int j = 0; j < TILES; ++j)
      update_h(h2, hb2, lane, nbase + 16 * j, acc[j], bt2[j]);
    __syncthreads();
  }

  // ---- write final states: out_h layout [3][BATCH][N_DIM] f32
  for (int i = tid; i < 3 * BROWS * N_DIM; i += NTHREADS) {
    const int l = i / (BROWS * N_DIM);
    const int r = (i / N_DIM) % BROWS;
    const int n = i % N_DIM;
    const float* hp = (l == 0) ? h0 : (l == 1) ? h1 : h2;
    out_h[((size_t)l * BATCH + b0 + r) * N_DIM + n] = hp[r * LDSF + n];
  }

  // ---- readout: out = h2 @ W_fc^T + b_fc  (tiny: 16 rows x 10 classes)
  if (tid < BROWS * N_CLASSES) {
    const int r = tid / N_CLASSES;
    const int c = tid - r * N_CLASSES;
    const float* wrow = W_fc + (size_t)c * N_DIM;
    float acc2 = b_fc[c];
    for (int n = 0; n < N_DIM; ++n) acc2 += h2[r * LDSF + n] * wrow[n];
    out_fc[(size_t)(b0 + r) * N_CLASSES + c] = acc2;
  }
}

// ---------------------------------------------------------------------------
extern "C" void kernel_launch(void* const* d_in, const int* in_sizes, int n_in,
                              void* d_out, int out_size, void* d_ws, size_t ws_size,
                              hipStream_t stream) {
  const float* data  = (const float*)d_in[0];
  const float* W_in  = (const float*)d_in[1];
  const float* b_in  = (const float*)d_in[2];
  const float* W_hh0 = (const float*)d_in[3];
  const float* b_hh0 = (const float*)d_in[4];
  const float* W_hh1 = (const float*)d_in[5];
  const float* b_hh1 = (const float*)d_in[6];
  const float* W_hh2 = (const float*)d_in[7];
  const float* b_hh2 = (const float*)d_in[8];
  const float* W_hi0 = (const float*)d_in[9];
  const float* b_hi0 = (const float*)d_in[10];
  const float* W_hi1 = (const float*)d_in[11];
  const float* b_hi1 = (const float*)d_in[12];
  const float* W_fc  = (const float*)d_in[13];
  const float* b_fc  = (const float*)d_in[14];

  unsigned short* wbf = (unsigned short*)d_ws;  // bf16 weights in workspace

  // 1) weight conversion (f32 -> bf16, zero recurrent diagonals)
  const long total = 5L * N_DIM * N_DIM + (long)N_DIM * IN_DIM;
  const int tpb = 256;
  const int nb  = (int)((total + tpb - 1) / tpb);
  prep_weights_bf16<<<nb, tpb, 0, stream>>>(W_hh0, W_hh1, W_hh2, W_hi0, W_hi1,
                                            W_in, wbf);

  // 2) persistent recurrence kernel
  float* out_h  = (float*)d_out;
  float* out_fc = out_h + 3 * BATCH * N_DIM;

  const size_t lds_bytes = (size_t)3 * BROWS * LDSF * sizeof(float) +
                           (size_t)3 * BROWS * LDSB * sizeof(unsigned short); // ~289.5 KB
  hipFuncSetAttribute((const void*)rnn_stack_persist,
                      hipFuncAttributeMaxDynamicSharedMemorySize, (int)lds_bytes);
  rnn_stack_persist<<<BATCH / BROWS, NTHREADS, lds_bytes, stream>>>(
      data, wbf, b_in, b_hh0, b_hh1, b_hh2, b_hi0, b_hi1, W_fc, b_fc,
      out_h, out_fc);
}
